// ContextEnhancedConvModule_14001593385703
// MI455X (gfx1250) — matrix operand order repaired and checked by reading, not compiled, not run on hardware
//
#include <hip/hip_runtime.h>
#include <hip/hip_bf16.h>

// ---- problem constants (match reference) ----
#define BATCH   16
#define IN_CH   1024
#define HID     256
#define GROUPS  8
#define CPG     32        // HID / GROUPS
#define HSZ     64
#define WSZ     64
#define HW      4096      // HSZ*WSZ
#define EPS     1e-5f

typedef __attribute__((ext_vector_type(16))) __bf16 v16bf;
typedef __attribute__((ext_vector_type(8)))  __bf16 v8bf;
typedef __attribute__((ext_vector_type(8)))  float  v8f;

// ---------------------------------------------------------------------------
// CDNA5 async global->LDS copy (one dword per lane), tracked by ASYNCcnt.
// Probe result: builtin expects (global int*, lds int*, imm offset, imm cpol).
// ---------------------------------------------------------------------------
typedef __attribute__((address_space(1))) int g_int;
typedef __attribute__((address_space(3))) int l_int;

__device__ __forceinline__ void async_copy_b32(const float* gsrc, float* lds_dst) {
#if __has_builtin(__builtin_amdgcn_global_load_async_to_lds_b32)
    __builtin_amdgcn_global_load_async_to_lds_b32(
        (g_int*)(void*)const_cast<float*>(gsrc),
        (l_int*)(void*)lds_dst,
        0, 0);
#else
    // generic LDS pointer: low 32 bits are the LDS offset (flat LDS aperture)
    const unsigned lds_off = (unsigned)(size_t)lds_dst;
    const unsigned long long ga = (unsigned long long)(size_t)gsrc;
    asm volatile("global_load_async_to_lds_b32 %0, %1, off"
                 :: "v"(lds_off), "v"(ga) : "memory");
#endif
}

__device__ __forceinline__ void wait_async0() {
#if __has_builtin(__builtin_amdgcn_s_wait_asynccnt)
    __builtin_amdgcn_s_wait_asynccnt(0);
#else
    asm volatile("s_wait_asynccnt 0x0" ::: "memory");
#endif
}

// ---------------------------------------------------------------------------
// Fragment loaders for v_wmma_f32_16x16x32_bf16 (wave32).
// A (16x32 bf16): lane L holds row m=L&15; half g=L>>4 holds K = g*16+e.
// Source rows are f32, converted to bf16.
// ---------------------------------------------------------------------------
__device__ __forceinline__ v16bf load_a_frag_f32(const float* __restrict__ row, int g) {
    const float4 lo0 = *(const float4*)(row + g * 16);
    const float4 lo1 = *(const float4*)(row + g * 16 + 4);
    const float4 hi0 = *(const float4*)(row + g * 16 + 8);
    const float4 hi1 = *(const float4*)(row + g * 16 + 12);
    v16bf a;
    a[0]  = (__bf16)lo0.x; a[1]  = (__bf16)lo0.y; a[2]  = (__bf16)lo0.z; a[3]  = (__bf16)lo0.w;
    a[4]  = (__bf16)lo1.x; a[5]  = (__bf16)lo1.y; a[6]  = (__bf16)lo1.z; a[7]  = (__bf16)lo1.w;
    a[8]  = (__bf16)hi0.x; a[9]  = (__bf16)hi0.y; a[10] = (__bf16)hi0.z; a[11] = (__bf16)hi0.w;
    a[12] = (__bf16)hi1.x; a[13] = (__bf16)hi1.y; a[14] = (__bf16)hi1.z; a[15] = (__bf16)hi1.w;
    return a;
}

// B (32x16 bf16): lane L holds col n=L&15; half g=L>>4 holds K = g*16+e.
// p points at 16 contiguous f32 (this lane's K-run) in LDS; converts to bf16.
__device__ __forceinline__ v16bf load_b_frag_lds_f32(const float* p) {
    const float4 f0 = *(const float4*)(p);
    const float4 f1 = *(const float4*)(p + 4);
    const float4 f2 = *(const float4*)(p + 8);
    const float4 f3 = *(const float4*)(p + 12);
    v16bf v;
    v[0]  = (__bf16)f0.x; v[1]  = (__bf16)f0.y; v[2]  = (__bf16)f0.z; v[3]  = (__bf16)f0.w;
    v[4]  = (__bf16)f1.x; v[5]  = (__bf16)f1.y; v[6]  = (__bf16)f1.z; v[7]  = (__bf16)f1.w;
    v[8]  = (__bf16)f2.x; v[9]  = (__bf16)f2.y; v[10] = (__bf16)f2.z; v[11] = (__bf16)f2.w;
    v[12] = (__bf16)f3.x; v[13] = (__bf16)f3.y; v[14] = (__bf16)f3.z; v[15] = (__bf16)f3.w;
    return v;
}

// ---------------------------------------------------------------------------
// 1) global average pool:  ctx_mean[b][c] = mean over HW of x[b][c]
// ---------------------------------------------------------------------------
__global__ __launch_bounds__(256) void k_pool(const float* __restrict__ x,
                                              float* __restrict__ ctx_mean) {
    const int bc = blockIdx.x;                   // b*IN_CH + c
    const float* p = x + (size_t)bc * HW;
    float s = 0.f;
    for (int n = threadIdx.x; n < HW; n += 256) s += p[n];
    __shared__ float red[256];
    red[threadIdx.x] = s;
    __syncthreads();
#pragma unroll
    for (int off = 128; off > 0; off >>= 1) {
        if (threadIdx.x < off) red[threadIdx.x] += red[threadIdx.x + off];
        __syncthreads();
    }
    if (threadIdx.x == 0) ctx_mean[bc] = red[0] * (1.0f / HW);
}

// ---------------------------------------------------------------------------
// 2-4) tiny dense layers: out[b][o] = act(dot(in[b], w[o]) + bias[o])
// ---------------------------------------------------------------------------
__global__ void k_fc(const float* __restrict__ in, const float* __restrict__ w,
                     const float* __restrict__ bias, float* __restrict__ out,
                     int In, int Out, int do_relu) {
    const int b = blockIdx.y;
    const int o = blockIdx.x * blockDim.x + threadIdx.x;
    if (o >= Out) return;
    const float* iv = in + (size_t)b * In;
    const float* wr = w + (size_t)o * In;
    float s = bias[o];
    for (int k = 0; k < In; ++k) s += iv[k] * wr[k];
    out[(size_t)b * Out + o] = do_relu ? fmaxf(s, 0.f) : s;
}

// ---------------------------------------------------------------------------
// 5) GEMM1: xt[b] = in_w(256x1024) @ x[b](1024x4096) + in_b   (bf16 WMMA)
// block: 256 thr = 8 waves; wave w -> rows [w*32, w*32+32); block -> 32 cols.
// x K-tile staged via async global->LDS with element transpose [k][n]->[n][k].
// ---------------------------------------------------------------------------
#define XT_STRIDE 36   // dword row stride: 16B aligned, bank-conflict-free
__global__ __launch_bounds__(256) void k_gemm_in(const float* __restrict__ x,
                                                 const float* __restrict__ in_w,
                                                 const float* __restrict__ in_b,
                                                 float* __restrict__ xt) {
    __shared__ __attribute__((aligned(16))) float xtile[32][XT_STRIDE];
    const int n0   = blockIdx.x * 32;            // spatial tile
    const int b    = blockIdx.y;
    const int tid  = threadIdx.x;
    const int wave = tid >> 5;
    const int lane = tid & 31;
    const int g    = lane >> 4;
    const int l15  = lane & 15;
    const int m_base = wave * 32;

    v8f acc[2][2] = {};
    const float* a_row0 = in_w + (size_t)(m_base + l15) * IN_CH;
    const float* a_row1 = in_w + (size_t)(m_base + 16 + l15) * IN_CH;
    const float* xb     = x + (size_t)b * IN_CH * HW + n0;

    for (int kk = 0; kk < IN_CH; kk += 32) {
        // ---- async transposing copy of the 32(K) x 32(N) f32 tile into LDS
#pragma unroll
        for (int i = 0; i < 4; ++i) {
            const int idx = tid + 256 * i;       // 0..1023
            const int k = idx >> 5;              // 0..31
            const int n = idx & 31;              // 0..31 (fastest -> coalesced)
            async_copy_b32(xb + (size_t)(kk + k) * HW + n, &xtile[n][k]);
        }
        if (kk + 32 < IN_CH)                     // global_prefetch_b8 next K chunk
            __builtin_prefetch(xb + (size_t)(kk + 32 + (tid >> 5)) * HW + (tid & 31), 0, 1);
        wait_async0();
        __syncthreads();

        const v16bf a0 = load_a_frag_f32(a_row0 + kk, g);
        const v16bf a1 = load_a_frag_f32(a_row1 + kk, g);
        const v16bf b0 = load_b_frag_lds_f32(&xtile[l15][g * 16]);
        const v16bf b1 = load_b_frag_lds_f32(&xtile[16 + l15][g * 16]);

        acc[0][0] = __builtin_amdgcn_wmma_f32_16x16x32_bf16(false, a0, false, b0, (short)0, acc[0][0], false, false);
        acc[0][1] = __builtin_amdgcn_wmma_f32_16x16x32_bf16(false, a0, false, b1, (short)0, acc[0][1], false, false);
        acc[1][0] = __builtin_amdgcn_wmma_f32_16x16x32_bf16(false, a1, false, b0, (short)0, acc[1][0], false, false);
        acc[1][1] = __builtin_amdgcn_wmma_f32_16x16x32_bf16(false, a1, false, b1, (short)0, acc[1][1], false, false);
        __syncthreads();                         // protect LDS tile before next copy
    }
    // C layout: VGPR r -> M = r + 8*g ; N = lane&15
#pragma unroll
    for (int i = 0; i < 2; ++i)
#pragma unroll
        for (int r = 0; r < 8; ++r) {
            const int row  = m_base + i * 16 + r + 8 * g;
            const float bia = in_b[row];
#pragma unroll
            for (int j = 0; j < 2; ++j) {
                const int col = n0 + j * 16 + l15;
                xt[((size_t)b * HID + row) * HW + col] = acc[i][j][r] + bia;
            }
        }
}

// ---------------------------------------------------------------------------
// 6) fused depthwise-3x3 + GEMM2:
//    dw tile (256 ch x 32 pos) built once in LDS (bf16), then
//    out2[b] = out_w(256x256) @ dw(256x4096) + out_b  via bf16 WMMA, K=256.
// ---------------------------------------------------------------------------
__global__ __launch_bounds__(256) void k_gemm_out(const float* __restrict__ xt,
                                                  const float* __restrict__ out_w,
                                                  const float* __restrict__ out_b,
                                                  const float* __restrict__ kern,
                                                  float* __restrict__ out2) {
    __shared__ __attribute__((aligned(16))) __bf16 dw[32][HID + 8]; // [n_local][k]
    const int n0 = blockIdx.x * 32;
    const int b  = blockIdx.y;

    // ---- phase 1: depthwise 3x3 into LDS (each thread: 1 column, 32 channels of 1 group)
    {
        const int nl  = threadIdx.x >> 3;        // 0..31 local spatial pos
        const int grp = threadIdx.x & 7;         // group id (k / 32)
        const int gn  = n0 + nl;
        const int h = gn >> 6, w = gn & 63;
        float kc[9];
#pragma unroll
        for (int t = 0; t < 9; ++t) kc[t] = kern[b * (GROUPS * 9) + grp * 9 + t];
        const float* xb = xt + ((size_t)b * HID + grp * CPG) * HW;
        for (int kk = 0; kk < CPG; ++kk) {
            const float* xc = xb + (size_t)kk * HW;
            float acc = 0.f;
#pragma unroll
            for (int dy = -1; dy <= 1; ++dy) {
                const int hh = h + dy;
                if (hh < 0 || hh >= HSZ) continue;
#pragma unroll
                for (int dx = -1; dx <= 1; ++dx) {
                    const int ww = w + dx;
                    if (ww < 0 || ww >= WSZ) continue;
                    acc += kc[(dy + 1) * 3 + (dx + 1)] * xc[hh * WSZ + ww];
                }
            }
            dw[nl][grp * CPG + kk] = (__bf16)acc;
        }
    }
    __syncthreads();

    // ---- phase 2: WMMA GEMM, A = out_w (global f32 -> bf16), B = LDS dw tile
    const int wave = threadIdx.x >> 5;
    const int lane = threadIdx.x & 31;
    const int g    = lane >> 4;
    const int l15  = lane & 15;
    const int m_base = wave * 32;

    v8f acc[2][2] = {};
    const float* a_row0 = out_w + (size_t)(m_base + l15) * HID;
    const float* a_row1 = out_w + (size_t)(m_base + 16 + l15) * HID;

    for (int kk = 0; kk < HID; kk += 32) {
        const v16bf a0 = load_a_frag_f32(a_row0 + kk, g);
        const v16bf a1 = load_a_frag_f32(a_row1 + kk, g);
        // B fragment from LDS: lane col n = sub*16 + l15, K = kk + g*16 + e (contiguous)
        const v8bf* p0 = (const v8bf*)&dw[l15][kk + g * 16];
        const v8bf* p1 = (const v8bf*)&dw[16 + l15][kk + g * 16];
        const v16bf b0 = __builtin_shufflevector(p0[0], p0[1],
                           0,1,2,3,4,5,6,7,8,9,10,11,12,13,14,15);
        const v16bf b1 = __builtin_shufflevector(p1[0], p1[1],
                           0,1,2,3,4,5,6,7,8,9,10,11,12,13,14,15);
        acc[0][0] = __builtin_amdgcn_wmma_f32_16x16x32_bf16(false, a0, false, b0, (short)0, acc[0][0], false, false);
        acc[0][1] = __builtin_amdgcn_wmma_f32_16x16x32_bf16(false, a0, false, b1, (short)0, acc[0][1], false, false);
        acc[1][0] = __builtin_amdgcn_wmma_f32_16x16x32_bf16(false, a1, false, b0, (short)0, acc[1][0], false, false);
        acc[1][1] = __builtin_amdgcn_wmma_f32_16x16x32_bf16(false, a1, false, b1, (short)0, acc[1][1], false, false);
    }
#pragma unroll
    for (int i = 0; i < 2; ++i)
#pragma unroll
        for (int r = 0; r < 8; ++r) {
            const int row  = m_base + i * 16 + r + 8 * g;
            const float bia = out_b[row];
#pragma unroll
            for (int j = 0; j < 2; ++j) {
                const int col = n0 + j * 16 + l15;
                out2[((size_t)b * HID + row) * HW + col] = acc[i][j][r] + bia;
            }
        }
}

// ---------------------------------------------------------------------------
// 7) batch-norm statistics: one block per channel, reduce over (B, HW)
// ---------------------------------------------------------------------------
__global__ __launch_bounds__(256) void k_bn_stats(const float* __restrict__ out2,
                                                  float* __restrict__ meanv,
                                                  float* __restrict__ varv) {
    const int c = blockIdx.x;
    float s = 0.f, sq = 0.f;
    for (int b = 0; b < BATCH; ++b) {
        const float* p = out2 + ((size_t)b * HID + c) * HW;
        for (int n = threadIdx.x; n < HW; n += 256) {
            const float v = p[n];
            s += v; sq += v * v;
        }
    }
    __shared__ float rs[256], rq[256];
    rs[threadIdx.x] = s; rq[threadIdx.x] = sq;
    __syncthreads();
#pragma unroll
    for (int off = 128; off > 0; off >>= 1) {
        if (threadIdx.x < off) {
            rs[threadIdx.x] += rs[threadIdx.x + off];
            rq[threadIdx.x] += rq[threadIdx.x + off];
        }
        __syncthreads();
    }
    if (threadIdx.x == 0) {
        const float inv_n = 1.0f / (float)(BATCH * HW);
        const float m = rs[0] * inv_n;
        meanv[c] = m;
        varv[c]  = rq[0] * inv_n - m * m;
    }
}

// ---------------------------------------------------------------------------
// 8) final: out = xt + gamma*(out2 - mean)*rsqrt(var+eps) + beta
// ---------------------------------------------------------------------------
__global__ __launch_bounds__(256) void k_bn_final(const float* __restrict__ xt,
                                                  const float* __restrict__ out2,
                                                  const float* __restrict__ meanv,
                                                  const float* __restrict__ varv,
                                                  const float* __restrict__ gamma,
                                                  const float* __restrict__ beta,
                                                  float* __restrict__ out) {
    const size_t idx = (size_t)blockIdx.x * 256 + threadIdx.x;
    const int c = (int)((idx >> 12) & (HID - 1));   // HW == 4096 == 1<<12
    const float inv = rsqrtf(varv[c] + EPS);
    out[idx] = xt[idx] + gamma[c] * (out2[idx] - meanv[c]) * inv + beta[c];
}

// ---------------------------------------------------------------------------
extern "C" void kernel_launch(void* const* d_in, const int* in_sizes, int n_in,
                              void* d_out, int out_size, void* d_ws, size_t ws_size,
                              hipStream_t stream) {
    const float* x     = (const float*)d_in[0];
    const float* ctx_w = (const float*)d_in[1];
    const float* ctx_b = (const float*)d_in[2];
    const float* kg_w1 = (const float*)d_in[3];
    const float* kg_b1 = (const float*)d_in[4];
    const float* kg_w2 = (const float*)d_in[5];
    const float* kg_b2 = (const float*)d_in[6];
    const float* in_w  = (const float*)d_in[7];
    const float* in_b  = (const float*)d_in[8];
    const float* out_w = (const float*)d_in[9];
    const float* out_b = (const float*)d_in[10];
    const float* gamma = (const float*)d_in[11];
    const float* beta  = (const float*)d_in[12];
    float* out = (float*)d_out;

    // workspace layout (floats)
    float* ws       = (float*)d_ws;
    float* ctx_mean = ws;                                    // 16*1024
    float* ctxh     = ctx_mean + BATCH * IN_CH;              // 16*256
    float* hddn     = ctxh + BATCH * HID;                    // 16*512
    float* kern     = hddn + BATCH * 2 * HID;                // 16*72
    float* meanv    = kern + BATCH * GROUPS * 9;             // 256
    float* varv     = meanv + HID;                           // 256
    float* xt       = varv + HID;                            // 16*256*4096
    float* out2     = xt + (size_t)BATCH * HID * HW;         // 16*256*4096

    // 1) context pooling
    k_pool<<<BATCH * IN_CH, 256, 0, stream>>>(x, ctx_mean);
    // 2) ctx 1x1 conv + relu: [16,1024] -> [16,256]
    k_fc<<<dim3(1, BATCH), 256, 0, stream>>>(ctx_mean, ctx_w, ctx_b, ctxh, IN_CH, HID, 1);
    // 3) kg layer 1 + relu: [16,256] -> [16,512]
    k_fc<<<dim3(2, BATCH), 256, 0, stream>>>(ctxh, kg_w1, kg_b1, hddn, HID, 2 * HID, 1);
    // 4) kg layer 2: [16,512] -> [16,72]
    k_fc<<<dim3(1, BATCH), 128, 0, stream>>>(hddn, kg_w2, kg_b2, kern, 2 * HID, GROUPS * 9, 0);
    // 5) input transform GEMM (WMMA bf16, async-LDS staged B operand)
    k_gemm_in<<<dim3(HW / 32, BATCH), 256, 0, stream>>>(x, in_w, in_b, xt);
    // 6) fused depthwise conv + output transform GEMM (WMMA bf16)
    k_gemm_out<<<dim3(HW / 32, BATCH), 256, 0, stream>>>(xt, out_w, out_b, kern, out2);
    // 7) batch statistics
    k_bn_stats<<<HID, 256, 0, stream>>>(out2, meanv, varv);
    // 8) BN + residual
    k_bn_final<<<(BATCH * HID * HW) / 256, 256, 0, stream>>>(xt, out2, meanv, varv,
                                                             gamma, beta, out);
}
